// Attention_84851373900044
// MI455X (gfx1250) — matrix operand order repaired and checked by reading, not compiled, not run on hardware
//
#include <hip/hip_runtime.h>
#include <hip/hip_bf16.h>

// MI455X / gfx1250 fused attention:
//   qkv = W @ x  (bf16 WMMA, f32 accum, LDS-staged x panel)
//   logits = [Q;P]^T [K;Q]  -> 16x1024 f32 score panel lives in 320KB LDS
//   out = V @ A^T           (bf16 WMMA, f32 accum, 1/rowsum folded into store)

typedef __attribute__((ext_vector_type(16))) __bf16 bf16x16;
typedef __attribute__((ext_vector_type(8)))  __bf16 bf16x8;
typedef __attribute__((ext_vector_type(8)))  float  f32x8;
typedef __attribute__((ext_vector_type(4)))  float  f32x4;

#define B_DIM 64
#define C_DIM 512
#define N_DIM 1024
#define O_DIM 1536   // 3*C

__device__ __forceinline__ f32x8 wmma_bf16(bf16x16 a, bf16x16 b, f32x8 c) {
  // (neg_a, A, neg_b, B, c_mod, C, reuse_a, reuse_b)
  return __builtin_amdgcn_wmma_f32_16x16x32_bf16(false, a, false, b,
                                                 (short)0, c, false, false);
}

// A-matrix fragment (16x32 bf16), row-major source, row stride `stride`.
// lane<16 : row=lane,    K in {0..7, 16..23}
// lane>=16: row=lane-16, K in {8..15, 24..31}
__device__ __forceinline__ bf16x16 frag_a(const __bf16* rowbase, int stride) {
  const int lane = threadIdx.x & 31;
  const int row  = lane & 15;
  const int koff = (lane >> 4) << 3;          // 0 or 8
  const __bf16* p = rowbase + row * stride + koff;
  bf16x8 lo = *(const bf16x8*)(p);            // K = koff .. koff+7
  bf16x8 hi = *(const bf16x8*)(p + 16);       // K = 16+koff .. 16+koff+7
  bf16x16 r;
#pragma unroll
  for (int i = 0; i < 8; ++i) { r[i] = lo[i]; r[i + 8] = hi[i]; }
  return r;
}

// B-matrix fragment (32x16 bf16). Source holds B^T row-major:
// row (lane&15) = N column of B; K runs contiguously along the row.
// lane<16: K in {0..7, 16..23}, lane>=16: K in {8..15, 24..31}.
__device__ __forceinline__ bf16x16 frag_b(const __bf16* rowbase, int stride) {
  const int lane = threadIdx.x & 31;
  const int col  = lane & 15;
  const int kh   = (lane >> 4) << 3;          // 0 or 8
  const __bf16* p = rowbase + col * stride + kh;
  bf16x8 lo = *(const bf16x8*)(p);
  bf16x8 hi = *(const bf16x8*)(p + 16);
  bf16x16 r;
#pragma unroll
  for (int i = 0; i < 8; ++i) { r[i] = lo[i]; r[i + 8] = hi[i]; }
  return r;
}

__global__ void prep_w(const float* __restrict__ W, __bf16* __restrict__ Wb) {
  int i = blockIdx.x * 256 + threadIdx.x;
  if (i < O_DIM * C_DIM) Wb[i] = (__bf16)W[i];
}

// Pt[n][c] = rel_h[c][h] + rel_w[c][w],  n = h*32 + w   (transposed for A-frags)
__global__ void prep_pos(const float* __restrict__ rel_h,
                         const float* __restrict__ rel_w,
                         __bf16* __restrict__ Pt) {
  int i = blockIdx.x * 256 + threadIdx.x;
  if (i < N_DIM * C_DIM) {
    int n = i >> 9, c = i & 511;
    int h = n >> 5, w = n & 31;
    Pt[i] = (__bf16)(rel_h[c * 32 + h] + rel_w[c * 32 + w]);
  }
}

// One block per (b, 16-wide n-tile). Stage x panel in LDS as bf16,
// 8 waves sweep the 96 output row-tiles of W in 2 groups of 6 accumulators
// (B-fragment hoisted out of the tile loop: one ds read feeds 6 WMMAs).
__global__ __launch_bounds__(256) void qkv_kernel(
    const float* __restrict__ x, const __bf16* __restrict__ Wb,
    __bf16* __restrict__ Qt, __bf16* __restrict__ Kt, __bf16* __restrict__ V) {
  __shared__ __bf16 Xl[16 * C_DIM];           // [n_local][c], 16 KB

  const int b   = blockIdx.x >> 6;
  const int n0  = (blockIdx.x & 63) << 4;
  const int tid = threadIdx.x;
  const float* xb = x + (size_t)b * C_DIM * N_DIM;

  // 512 rows x 16 cols, b128 loads (n0 is 64B aligned)
#pragma unroll
  for (int rep = 0; rep < 2; ++rep) {
    const int c = tid + (rep << 8);
    const float* src = xb + (size_t)c * N_DIM + n0;
#pragma unroll
    for (int q = 0; q < 4; ++q) {
      f32x4 f = *(const f32x4*)(src + q * 4);
#pragma unroll
      for (int j = 0; j < 4; ++j)
        Xl[(q * 4 + j) * C_DIM + c] = (__bf16)f[j];
    }
  }
  __syncthreads();

  const int wave = tid >> 5;
  const int lane = tid & 31;
  for (int g = 0; g < 2; ++g) {
    f32x8 acc[6] = {};
    for (int c0 = 0; c0 < C_DIM; c0 += 32) {
      bf16x16 bx = frag_b(&Xl[c0], C_DIM);          // x cols (LDS), shared x6
#pragma unroll
      for (int t = 0; t < 6; ++t) {
        const int ot = wave + ((g * 6 + t) << 3);   // 0..95
        bf16x16 a = frag_a(Wb + (size_t)(ot << 4) * C_DIM + c0, C_DIM);
        acc[t] = wmma_bf16(a, bx, acc[t]);
      }
    }
    // D layout: lane<16 -> (M=r, N=lane), lane>=16 -> (M=r+8, N=lane-16)
#pragma unroll
    for (int t = 0; t < 6; ++t) {
      const int ot = wave + ((g * 6 + t) << 3);
      const int ng = n0 + (lane & 15);
      const int ob = (ot << 4) + ((lane >> 4) << 3);  // first of 8 consecutive o
      if (ot < 32) {                                  // Q -> Qt[b][n][c]
        bf16x8 v8;
#pragma unroll
        for (int r = 0; r < 8; ++r) v8[r] = (__bf16)acc[t][r];
        *(bf16x8*)(Qt + ((size_t)b * N_DIM + ng) * C_DIM + ob) = v8;
      } else if (ot < 64) {                           // K -> Kt[b][n][c]
        bf16x8 v8;
#pragma unroll
        for (int r = 0; r < 8; ++r) v8[r] = (__bf16)acc[t][r];
        *(bf16x8*)(Kt + ((size_t)b * N_DIM + ng) * C_DIM + (ob - 512)) = v8;
      } else {                                        // V -> V[b][c][n]
        const int cb = ob - 1024;
#pragma unroll
        for (int r = 0; r < 8; ++r)
          V[((size_t)b * C_DIM + cb + r) * N_DIM + ng] = (__bf16)acc[t][r];
      }
    }
  }
}

// One block per (b, 16-row i-tile): full score panel + softmax in LDS, then PV.
__global__ __launch_bounds__(256) void attn_kernel(
    const __bf16* __restrict__ Qt, const __bf16* __restrict__ Kt,
    const __bf16* __restrict__ V,  const __bf16* __restrict__ Pt,
    float* __restrict__ out) {
  __shared__ float  S[16 * N_DIM];      // 64 KB score panel
  __shared__ __bf16 Pr[16 * N_DIM];     // 32 KB bf16 probabilities
  __shared__ float  red[16 * 16];
  __shared__ float  rowmax[16];
  __shared__ float  rinv[16];

  const int b    = blockIdx.x >> 6;
  const int i0   = (blockIdx.x & 63) << 4;
  const int tid  = threadIdx.x;
  const int wave = tid >> 5, lane = tid & 31;

  const __bf16* Qb = Qt + (size_t)b * N_DIM * C_DIM;
  const __bf16* Kb = Kt + (size_t)b * N_DIM * C_DIM;
  const __bf16* Vb = V  + (size_t)b * C_DIM * N_DIM;
  const __bf16* Qi = Qb + (size_t)i0 * C_DIM;   // A rows for content term
  const __bf16* Pi = Pt + (size_t)i0 * C_DIM;   // A rows for position term

  // ---- scores: S[i][j] = sum_c Q[c,i]K[c,j] + P[c,i]Q[c,j] ----
  // c0 outer so one (A_q, A_p) fragment pair feeds all 8 j-tiles (16 WMMAs).
  {
    f32x8 acc[8] = {};
    for (int c0 = 0; c0 < C_DIM; c0 += 32) {
      bf16x16 aq = frag_a(Qi + c0, C_DIM);
      bf16x16 ap = frag_a(Pi + c0, C_DIM);
#pragma unroll
      for (int t = 0; t < 8; ++t) {
        const size_t j0 = (size_t)(((wave << 3) + t) << 4);
        bf16x16 bk = frag_b(Kb + j0 * C_DIM + c0, C_DIM);
        acc[t] = wmma_bf16(aq, bk, acc[t]);
        bf16x16 bq = frag_b(Qb + j0 * C_DIM + c0, C_DIM);
        acc[t] = wmma_bf16(ap, bq, acc[t]);
      }
    }
    const int col0 = lane & 15;
    const int rb   = (lane >> 4) << 3;
#pragma unroll
    for (int t = 0; t < 8; ++t) {
      const int j0 = ((wave << 3) + t) << 4;
#pragma unroll
      for (int r = 0; r < 8; ++r) S[(rb + r) * N_DIM + j0 + col0] = acc[t][r];
    }
  }
  __syncthreads();

  // ---- softmax: 16 threads per row, 64 cols per thread ----
  {
    const int row = tid >> 4, seg = tid & 15;
    const float* sr = &S[row * N_DIM + seg * 64];
    float m = -3.0e38f;
    for (int k = 0; k < 64; ++k) m = fmaxf(m, sr[k]);
    red[row * 16 + seg] = m;
    __syncthreads();
    if (seg == 0) {
      float mm = red[row * 16];
      for (int k = 1; k < 16; ++k) mm = fmaxf(mm, red[row * 16 + k]);
      rowmax[row] = mm;
    }
    __syncthreads();
    const float mm = rowmax[row];
    float sum = 0.f;
    __bf16* pr = &Pr[row * N_DIM + seg * 64];
    for (int k = 0; k < 64; ++k) {
      float e = __expf(sr[k] - mm);   // v_exp_f32 co-executes with WMMA waves
      sum += e;
      pr[k] = (__bf16)e;
    }
    red[row * 16 + seg] = sum;
    __syncthreads();
    if (seg == 0) {
      float ss = 0.f;
      for (int k = 0; k < 16; ++k) ss += red[row * 16 + k];
      rinv[row] = 1.0f / ss;
    }
    __syncthreads();
  }

  // ---- PV: each wave owns a 64-channel slice; O[i][c] = sum_j Pr[i][j]V[c][j]
  const int cw = wave << 6;
  f32x8 o0 = {}, o1 = {}, o2 = {}, o3 = {};
  for (int j0 = 0; j0 < N_DIM; j0 += 32) {
    bf16x16 ap = frag_a(&Pr[j0], N_DIM);                        // probs (LDS)
    bf16x16 b0 = frag_b(Vb + (size_t)(cw)*N_DIM + j0, N_DIM);
    bf16x16 b1 = frag_b(Vb + (size_t)(cw + 16) * N_DIM + j0, N_DIM);
    bf16x16 b2 = frag_b(Vb + (size_t)(cw + 32) * N_DIM + j0, N_DIM);
    bf16x16 b3 = frag_b(Vb + (size_t)(cw + 48) * N_DIM + j0, N_DIM);
    o0 = wmma_bf16(ap, b0, o0);
    o1 = wmma_bf16(ap, b1, o1);
    o2 = wmma_bf16(ap, b2, o2);
    o3 = wmma_bf16(ap, b3, o3);
  }

  // out[b][c][i] with 1/rowsum folded in; 8 consecutive i per lane -> 2x b128.
  const int rb = (lane >> 4) << 3;
  const int cl = lane & 15;
  float rs[8];
#pragma unroll
  for (int r = 0; r < 8; ++r) rs[r] = rinv[rb + r];
  f32x8 os[4] = {o0, o1, o2, o3};
#pragma unroll
  for (int k = 0; k < 4; ++k) {
    float* dst = out + (size_t)b * C_DIM * N_DIM +
                 (size_t)(cw + (k << 4) + cl) * N_DIM + i0 + rb;
    f32x4 lo = {os[k][0] * rs[0], os[k][1] * rs[1],
                os[k][2] * rs[2], os[k][3] * rs[3]};
    f32x4 hi = {os[k][4] * rs[4], os[k][5] * rs[5],
                os[k][6] * rs[6], os[k][7] * rs[7]};
    *(f32x4*)(dst)     = lo;
    *(f32x4*)(dst + 4) = hi;
  }
}

extern "C" void kernel_launch(void* const* d_in, const int* in_sizes, int n_in,
                              void* d_out, int out_size, void* d_ws, size_t ws_size,
                              hipStream_t stream) {
  (void)in_sizes; (void)n_in; (void)out_size; (void)ws_size;
  const float* x     = (const float*)d_in[0];
  const float* W     = (const float*)d_in[1];
  const float* rel_h = (const float*)d_in[2];
  const float* rel_w = (const float*)d_in[3];
  float* out = (float*)d_out;

  char* ws = (char*)d_ws;
  __bf16* Wb = (__bf16*)ws; ws += (size_t)O_DIM * C_DIM * 2;          // 1.5 MB
  __bf16* Pt = (__bf16*)ws; ws += (size_t)N_DIM * C_DIM * 2;          // 1 MB
  __bf16* Qt = (__bf16*)ws; ws += (size_t)B_DIM * N_DIM * C_DIM * 2;  // 64 MB
  __bf16* Kt = (__bf16*)ws; ws += (size_t)B_DIM * N_DIM * C_DIM * 2;  // 64 MB
  __bf16* Vv = (__bf16*)ws;                                           // 64 MB

  prep_w  <<<(O_DIM * C_DIM + 255) / 256, 256, 0, stream>>>(W, Wb);
  prep_pos<<<(N_DIM * C_DIM + 255) / 256, 256, 0, stream>>>(rel_h, rel_w, Pt);
  qkv_kernel <<<B_DIM * 64, 256, 0, stream>>>(x, Wb, Qt, Kt, Vv);
  attn_kernel<<<B_DIM * 64, 256, 0, stream>>>(Qt, Kt, Vv, Pt, out);
}
